// Encoder_4793183502909
// MI455X (gfx1250) — matrix operand order repaired and checked by reading, not compiled, not run on hardware
//
#include <hip/hip_runtime.h>

typedef __attribute__((ext_vector_type(16))) _Float16 v16h;
typedef __attribute__((ext_vector_type(8)))  _Float16 v8h;
typedef __attribute__((ext_vector_type(8)))  float    v8f;

constexpr int kB  = 64;     // batch
constexpr int kS  = 512;    // seq len
constexpr int kU  = 256;    // units
constexpr int kNG = 768;    // 3*U (z,r,h gates)
constexpr int kMT = (kB * kS) / 16;  // 2048 M-tiles
constexpr int kNT = kNG / 16;        // 48 N-tiles
constexpr int kKF = kU / 32;         // 8 K-fragments of 32

__device__ __forceinline__ v8f wmma16(v16h a, v16h b, v8f c) {
    return __builtin_amdgcn_wmma_f32_16x16x32_f16(false, a, false, b, (short)0, c, false, false);
}

// A-fragment (16x32 f16, ISA 7.12.2): lane = (m&15) + 16*hi,
// half index idx=2v+half maps to kk = (v&3)*2 + (v>=4?16:0) + hi*8 + half
__device__ __forceinline__ int a_kk(int idx, int hi) {
    int v = idx >> 1, half = idx & 1;
    return ((v & 3) * 2) + ((v >= 4) ? 16 : 0) + hi * 8 + half;
}
// B-fragment (32x16 f16): same k order as A-fragment (kk = (v&3)*2 + (v>=4?16:0) + hi*8 + half)
__device__ __forceinline__ int b_kk(int idx, int hi) {
    int v = idx >> 1, half = idx & 1;
    return ((v & 3) * 2) + ((v >= 4) ? 16 : 0) + hi * 8 + half;
}

// ---------------- K0: pack recurrent/input weights into B-fragment layout ---
__global__ void pack_wu(const float* __restrict__ Wf, const float* __restrict__ Wb,
                        const float* __restrict__ Uf, const float* __restrict__ Ub,
                        _Float16* __restrict__ wpack, _Float16* __restrict__ upack) {
    int j   = blockIdx.x;   // n-tile (0..47)
    int mat = blockIdx.y;   // 0=Wf 1=Wb 2=Uf 3=Ub
    const float* src = (mat == 0) ? Wf : (mat == 1) ? Wb : (mat == 2) ? Uf : Ub;
    _Float16* dst = (mat < 2) ? wpack + (size_t)mat * kNT * kKF * 32 * 16
                              : upack + (size_t)(mat - 2) * kNT * kKF * 32 * 16;
    int t = threadIdx.x, kf = t >> 5, lane = t & 31;
    int n = lane & 15, hi = lane >> 4;
    _Float16* d = dst + ((size_t)(j * kKF + kf) * 32 + lane) * 16;
#pragma unroll
    for (int idx = 0; idx < 16; ++idx) {
        int k = kf * 32 + b_kk(idx, hi);
        d[idx] = (_Float16)src[(size_t)k * kNG + j * 16 + n];
    }
}

// ------- K1: embedding gather -> A-fragment layout (f16) + transposed mask --
__global__ void gather_pack_x(const int* __restrict__ x, const float* __restrict__ emb,
                              _Float16* __restrict__ xpack, unsigned char* __restrict__ mask_t) {
    int mt = blockIdx.x;
    int t = threadIdx.x, kf = t >> 5, lane = t & 31;
    int m = lane & 15, hi = lane >> 4;
    int row = mt * 16 + m;          // row = b*kS + s (flat)
    int tok = x[row];
    if (t < 16) {                   // mask_t[s*kB + b] = (token != 0)
        int r2 = mt * 16 + t;
        mask_t[(size_t)(r2 & (kS - 1)) * kB + (r2 >> 9)] = (x[r2] != 0) ? 1 : 0;
    }
    const float* e = emb + (size_t)tok * kU;
    _Float16* d = xpack + ((size_t)(mt * kKF + kf) * 32 + lane) * 16;
#pragma unroll
    for (int idx = 0; idx < 16; ++idx) {
        int k = kf * 32 + a_kk(idx, hi);
        d[idx] = (_Float16)e[k];
    }
}

// -------- K2: GX = X @ W + b_in, stored batch-transposed [dir][s][col][64] --
__global__ void __launch_bounds__(256) gemm_gx(const _Float16* __restrict__ xpack,
                                               const v16h* __restrict__ wpack,
                                               const float* __restrict__ bf,
                                               const float* __restrict__ bb,
                                               _Float16* __restrict__ gx2) {
    __shared__ _Float16 xtile[16 * kU];   // 8 KB, fragment-order A tile
    int mt  = blockIdx.x;
    int tid = threadIdx.x, w = tid >> 5, lane = tid & 31;
    int dir = w >> 2, ww = w & 3;         // 4 waves per direction

    // async DMA the 8KB A-tile into LDS (2 x b128 per thread)
    {
        uint32_t lbase = (uint32_t)(uintptr_t)(&xtile[0]);
        uint64_t gbase = (uint64_t)(uintptr_t)(xpack + (size_t)mt * kKF * 32 * 16);
#pragma unroll
        for (int it = 0; it < 2; ++it) {
            uint32_t off = (uint32_t)(it * 256 + tid) * 16u;
            uint32_t l = lbase + off;
            uint64_t g = gbase + off;
            asm volatile("global_load_async_to_lds_b128 %0, %1, off"
                         :: "v"(l), "v"(g) : "memory");
        }
        asm volatile("s_wait_asynccnt 0x0" ::: "memory");
    }
    __syncthreads();

    const v16h* xtv = (const v16h*)xtile;
    const float* bias = dir ? bb : bf;    // row 0 = b_in
    v8f acc[12] = {};
#pragma unroll
    for (int kf = 0; kf < kKF; ++kf) {
        v16h a = xtv[kf * 32 + lane];
        v16h b[12];
#pragma unroll
        for (int j = 0; j < 12; ++j)
            b[j] = wpack[(((size_t)dir * kNT + ww * 12 + j) * kKF + kf) * 32 + lane];
#pragma unroll
        for (int j = 0; j < 12; ++j)
            acc[j] = wmma16(a, b[j], acc[j]);
    }
    int n = lane & 15, hi = lane >> 4;
    int s0 = (mt * 16) & (kS - 1);        // tiles never cross batch rows
    int bi = (mt * 16) >> 9;
#pragma unroll
    for (int j = 0; j < 12; ++j) {
        int col = (ww * 12 + j) * 16 + n;
        float bv = bias[col];
#pragma unroll
        for (int i = 0; i < 8; ++i) {
            int s = s0 + hi * 8 + i;
            gx2[(((size_t)dir * kS + s) * kNG + col) * kB + bi] = (_Float16)(acc[j][i] + bv);
        }
    }
}

// ---------------- K3: sequential GRU scan, one block per (16 rows, dir) -----
// Wave w owns gate-matched N-tiles {w,w+8 | 16+w,24+w | 32+w,40+w}: gate math
// runs in-register on the WMMA accumulators; h state lives in registers; only
// the next-step A-fragments cross waves (double-buffered LDS, 1 barrier/step).
__global__ void __launch_bounds__(256, 1) gru_scan(const unsigned char* __restrict__ mask_t,
                                                   const v16h* __restrict__ upack,
                                                   const _Float16* __restrict__ gx2,
                                                   const float* __restrict__ bf,
                                                   const float* __restrict__ bb,
                                                   float* __restrict__ out) {
    __shared__ v16h hfrag[2][kKF * 32];   // 2 x 8 KB, h in A-fragment layout
    int rt = blockIdx.x, dir = blockIdx.y;
    int b0 = rt * 16;
    int tid = threadIdx.x, w = tid >> 5, lane = tid & 31;
    int n = lane & 15, hi = lane >> 4;
    const float* brec = (dir ? bb : bf) + kNG;           // row 1 = b_rec
    const v16h*  up   = upack + (size_t)dir * kNT * kKF * 32;

    int uc[2] = {16 * w + n, 128 + 16 * w + n};          // unit columns (c=0,1)
    float brz[2], brr[2], brh[2];
    int sk[2], sidx[2], shi2[2];                          // hfrag scatter coords
#pragma unroll
    for (int c = 0; c < 2; ++c) {
        int u = uc[c];
        brz[c] = brec[u]; brr[c] = brec[kU + u]; brh[c] = brec[2 * kU + u];
        int kk = u & 31;
        sk[c] = u >> 5; shi2[c] = (kk >> 3) & 1;
        int v = ((kk >> 4) & 1) * 4 + ((kk & 7) >> 1);
        sidx[c] = 2 * v + (kk & 1);
    }
    {   // zero h fragments for step 0
        _Float16* hf = (_Float16*)&hfrag[0][0];
        for (int i = tid; i < kKF * 32 * 16; i += 256) hf[i] = (_Float16)0.f;
    }
    float hp[2][8];
#pragma unroll
    for (int c = 0; c < 2; ++c)
#pragma unroll
        for (int i = 0; i < 8; ++i) hp[c][i] = 0.f;

    for (int t = 0; t < kS; ++t) {
        int s = dir ? (kS - 1 - t) : t;
        __syncthreads();  // previous step's fragment writes visible
        const v16h* bufR = &hfrag[t & 1][0];
        _Float16*   bufW = (_Float16*)&hfrag[(t + 1) & 1][0];

        // ---- WMMA: gh = h @ U for this wave's 6 gate-matched tiles ----
        v8f zacc[2] = {}, racc[2] = {}, hacc[2] = {};
#pragma unroll
        for (int kf = 0; kf < kKF; ++kf) {
            v16h a  = bufR[kf * 32 + lane];
            v16h bz0 = up[((size_t)(w)      * kKF + kf) * 32 + lane];
            v16h bz1 = up[((size_t)(w + 8)  * kKF + kf) * 32 + lane];
            v16h br0 = up[((size_t)(16 + w) * kKF + kf) * 32 + lane];
            v16h br1 = up[((size_t)(24 + w) * kKF + kf) * 32 + lane];
            v16h bh0 = up[((size_t)(32 + w) * kKF + kf) * 32 + lane];
            v16h bh1 = up[((size_t)(40 + w) * kKF + kf) * 32 + lane];
            zacc[0] = wmma16(a, bz0, zacc[0]);
            zacc[1] = wmma16(a, bz1, zacc[1]);
            racc[0] = wmma16(a, br0, racc[0]);
            racc[1] = wmma16(a, br1, racc[1]);
            hacc[0] = wmma16(a, bh0, hacc[0]);
            hacc[1] = wmma16(a, bh1, hacc[1]);
        }

        // ---- gx (batch-transposed) + mask loads ----
        const _Float16* gb = gx2 + ((size_t)(dir * kS + s) * kNG) * kB + b0 + hi * 8;
        v8h gz[2], gr[2], gh[2];
#pragma unroll
        for (int c = 0; c < 2; ++c) {
            gz[c] = __builtin_nontemporal_load((const v8h*)(gb + (size_t)uc[c] * kB));
            gr[c] = __builtin_nontemporal_load((const v8h*)(gb + (size_t)(kU + uc[c]) * kB));
            gh[c] = __builtin_nontemporal_load((const v8h*)(gb + (size_t)(2 * kU + uc[c]) * kB));
        }
        uint64_t km = *(const uint64_t*)(mask_t + (size_t)s * kB + b0 + hi * 8);
        int sn = dir ? (s - 1) : (s + 1);
        if (sn >= 0 && sn < kS) {
            const _Float16* pb = gx2 + ((size_t)(dir * kS + sn) * kNG) * kB + b0 + hi * 8;
#pragma unroll
            for (int c = 0; c < 2; ++c) __builtin_prefetch(pb + (size_t)uc[c] * kB, 0, 1);
        }

        // ---- in-register gate combine, state update, outputs ----
#pragma unroll
        for (int c = 0; c < 2; ++c) {
#pragma unroll
            for (int i = 0; i < 8; ++i) {
                float z = 1.f / (1.f + __expf(-((float)gz[c][i] + zacc[c][i] + brz[c])));
                float r = 1.f / (1.f + __expf(-((float)gr[c][i] + racc[c][i] + brr[c])));
                float pre = (float)gh[c][i] + r * (hacc[c][i] + brh[c]);
                float hh = 2.f / (1.f + __expf(-2.f * pre)) - 1.f;   // tanh
                float keep = (float)((km >> (8 * i)) & 1ull);
                float hn = z * hp[c][i] + (1.f - z) * hh;
                hn = keep * hn + (1.f - keep) * hp[c][i];            // masked: carry
                hp[c][i] = hn;
                int m = hi * 8 + i;
                bufW[(sk[c] * 32 + m + shi2[c] * 16) * 16 + sidx[c]] = (_Float16)hn;
                atomicAdd(&out[((size_t)(b0 + m) * kS + s) * kU + uc[c]], hn);
            }
        }
    }
}

extern "C" void kernel_launch(void* const* d_in, const int* in_sizes, int n_in,
                              void* d_out, int out_size, void* d_ws, size_t ws_size,
                              hipStream_t stream) {
    (void)in_sizes; (void)n_in; (void)ws_size;
    const int*   x   = (const int*)d_in[0];
    const float* emb = (const float*)d_in[1];
    const float* Wf  = (const float*)d_in[2];
    const float* Uf  = (const float*)d_in[3];
    const float* bf  = (const float*)d_in[4];
    const float* Wb  = (const float*)d_in[5];
    const float* Ub  = (const float*)d_in[6];
    const float* bb  = (const float*)d_in[7];
    float* out = (float*)d_out;

    char* ws = (char*)d_ws;
    _Float16*      xpack  = (_Float16*)ws;                              // 16 MB
    _Float16*      wpack  = (_Float16*)(ws + 16777216);                 // 768 KB
    _Float16*      upack  = (_Float16*)(ws + 16777216 + 786432);        // 768 KB
    unsigned char* mask_t = (unsigned char*)(ws + 16777216 + 2 * 786432);   // 32 KB
    _Float16*      gx2    = (_Float16*)(ws + 16777216 + 2 * 786432 + 32768); // 96 MB

    hipMemsetAsync(d_out, 0, (size_t)out_size * sizeof(float), stream);
    pack_wu<<<dim3(kNT, 4), 256, 0, stream>>>(Wf, Wb, Uf, Ub, wpack, upack);
    gather_pack_x<<<kMT, 256, 0, stream>>>(x, emb, xpack, mask_t);
    gemm_gx<<<kMT, 256, 0, stream>>>(xpack, (const v16h*)wpack, bf, bb, gx2);
    gru_scan<<<dim3(4, 2), 256, 0, stream>>>(mask_t, (const v16h*)upack, gx2, bf, bb, out);
}
